// Decoder_18811956756720
// MI455X (gfx1250) — compile-verified, matrix operand order checked by reading, not run on hardware
//
#include <hip/hip_runtime.h>
#include <hip/hip_bf16.h>
#include <stdint.h>

// ---------------------------------------------------------------------------
// Problem constants (from reference)
// ---------------------------------------------------------------------------
#define BATCH   2048
#define TSTEPS  256
#define HDIM    512          // ENC_DIM
#define NPARTS  64
#define PDIM    32
#define ADIM    128
#define VOCAB   12
#define KCAT    576          // 512 (h) + 44 (x_t|ctx) + 20 pad  -> 18 * 32
#define NUB     192          // 128 (Uh) + 32 (beta) + 32 pad    -> 3 * 64

typedef __attribute__((ext_vector_type(16))) __bf16 v16bf;
typedef __attribute__((ext_vector_type(8)))  float  v8f;

// ---------------------------------------------------------------------------
// WMMA bf16 fragment load (16-bit A/B 16x32 layout, CDNA5 ISA 7.12.2):
//   lanes 0-15 / 16-31 both hold rows M=0..15; per-lane halves are
//   VGPR0..3 -> K = (hi*8)+{0..7}, VGPR4..7 -> K = 16+(hi*8)+{0..7}
// Loaded as 8 dwords (two 16B-contiguous chunks -> compiler emits 2x b128).
// ---------------------------------------------------------------------------
__device__ __forceinline__ v16bf frag_load(const __hip_bfloat16* __restrict__ base,
                                           int row0, int ld, int k0, int r, int hi)
{
    union { uint32_t u[8]; v16bf v; } f;
    const uint32_t* p = reinterpret_cast<const uint32_t*>(base + (size_t)(row0 + r) * ld);
    const int kb = (k0 >> 1) + hi * 4;   // dword index of first K pair
#pragma unroll
    for (int v = 0; v < 4; ++v) f.u[v]     = p[kb + v];
#pragma unroll
    for (int v = 0; v < 4; ++v) f.u[4 + v] = p[kb + 8 + v];
    return f.v;
}

// D/C 16x16 f32 layout: VGPR i -> (M = hi*8 + i, N = r)
__device__ __forceinline__ void tile_store(float* __restrict__ C,
                                           const float* __restrict__ bias,
                                           v8f acc, int m0, int n0, int ldc,
                                           int r, int hi)
{
    const int col = n0 + r;
    const float bv = bias[col];
#pragma unroll
    for (int i = 0; i < 8; ++i) {
        const int row = m0 + hi * 8 + i;
        C[(size_t)row * ldc + col] = acc[i] + bv;
    }
}

// ---------------------------------------------------------------------------
// Generic NT bf16 GEMM:  C[M,N] = A[M,K] * B[N,K]^T + bias[N]
// Block = 256 threads = 8 waves arranged 4(M) x 2(N).
// Each wave owns a 64x32 macro-tile = 4x2 WMMA tiles (8 accumulators).
// Software-pipelined in registers: fragments for k+32 are loaded BEFORE the
// 8 v_wmma of step k issue, so WMMAs only ever wait on loads issued one full
// K-step earlier.  A reuse x2, B reuse x4 => ~22 FLOP per L2 byte; the whole
// per-step working set (<25 MB) is resident in the 192 MB L2.
// ---------------------------------------------------------------------------
__global__ __launch_bounds__(256)
void gemm_bf16_nt(const __hip_bfloat16* __restrict__ A,
                  const __hip_bfloat16* __restrict__ B,
                  const float* __restrict__ bias,
                  float* __restrict__ C,
                  int K, int lda, int ldb, int ldc)
{
    const int lane = threadIdx.x & 31;
    const int wave = threadIdx.x >> 5;
    const int r  = lane & 15;
    const int hi = lane >> 4;
    const int wm = wave >> 1;                      // 0..3
    const int wn = wave & 1;                       // 0..1
    const int m_base = blockIdx.x * 256 + wm * 64;
    const int n_base = blockIdx.y * 64  + wn * 32;

    v8f acc[4][2];
#pragma unroll
    for (int i = 0; i < 4; ++i)
#pragma unroll
        for (int j = 0; j < 2; ++j) acc[i][j] = (v8f){};

    // --- prologue: load fragments for k = 0 ---
    v16bf a[4], b[2];
#pragma unroll
    for (int i = 0; i < 4; ++i) a[i] = frag_load(A, m_base + 16 * i, lda, 0, r, hi);
#pragma unroll
    for (int j = 0; j < 2; ++j) b[j] = frag_load(B, n_base + 16 * j, ldb, 0, r, hi);

    // --- main loop: prefetch k+32, then consume k ---
    int k0 = 0;
#pragma unroll 2
    for (; k0 + 32 < K; k0 += 32) {
        v16bf an[4], bn[2];
#pragma unroll
        for (int i = 0; i < 4; ++i) an[i] = frag_load(A, m_base + 16 * i, lda, k0 + 32, r, hi);
#pragma unroll
        for (int j = 0; j < 2; ++j) bn[j] = frag_load(B, n_base + 16 * j, ldb, k0 + 32, r, hi);
#pragma unroll
        for (int i = 0; i < 4; ++i)
#pragma unroll
            for (int j = 0; j < 2; ++j)
                acc[i][j] = __builtin_amdgcn_wmma_f32_16x16x32_bf16(
                    false, a[i], false, b[j], (short)0, acc[i][j], false, false);
#pragma unroll
        for (int i = 0; i < 4; ++i) a[i] = an[i];
#pragma unroll
        for (int j = 0; j < 2; ++j) b[j] = bn[j];
    }

    // --- epilogue: last K-step ---
#pragma unroll
    for (int i = 0; i < 4; ++i)
#pragma unroll
        for (int j = 0; j < 2; ++j)
            acc[i][j] = __builtin_amdgcn_wmma_f32_16x16x32_bf16(
                false, a[i], false, b[j], (short)0, acc[i][j], false, false);

#pragma unroll
    for (int i = 0; i < 4; ++i)
#pragma unroll
        for (int j = 0; j < 2; ++j)
            tile_store(C, bias, acc[i][j], m_base + 16 * i, n_base + 16 * j, ldc, r, hi);
}

// ---------------------------------------------------------------------------
// One-time weight fusion + bf16 conversion.
//  Wcat[2048,576]  = [W_hh | W_ih | 0]     biasG[2048] = b_ih + b_hh
//  WUB [192,512]   = [Uw ; betaw ; 0]      biasU[192]  = [Ub ; betab ; 0]
// ---------------------------------------------------------------------------
__global__ void k_prep(const float* __restrict__ W_hh, const float* __restrict__ W_ih,
                       const float* __restrict__ Uw,   const float* __restrict__ betaw,
                       const float* __restrict__ b_ih, const float* __restrict__ b_hh,
                       const float* __restrict__ Ub,   const float* __restrict__ betab,
                       __hip_bfloat16* __restrict__ Wcat, __hip_bfloat16* __restrict__ WUB,
                       float* __restrict__ biasG, float* __restrict__ biasU)
{
    const int idx = blockIdx.x * 256 + threadIdx.x;
    const int NW = 2048 * KCAT;
    const int NU = NUB * HDIM;
    if (idx < NW) {
        const int n = idx / KCAT, k = idx % KCAT;
        float v = 0.f;
        if (k < HDIM)            v = W_hh[n * HDIM + k];
        else if (k < HDIM + 44)  v = W_ih[n * 44 + (k - HDIM)];
        Wcat[idx] = __float2bfloat16(v);
    } else if (idx < NW + NU) {
        const int j = idx - NW;
        const int n = j / HDIM, k = j % HDIM;
        float v = 0.f;
        if (n < ADIM)              v = Uw[n * HDIM + k];
        else if (n < ADIM + PDIM)  v = betaw[(n - ADIM) * HDIM + k];
        WUB[j] = __float2bfloat16(v);
    } else if (idx < NW + NU + 2048) {
        const int n = idx - NW - NU;
        biasG[n] = b_ih[n] + b_hh[n];
    } else if (idx < NW + NU + 2048 + NUB) {
        const int n = idx - NW - NU - 2048;
        float v = 0.f;
        if (n < ADIM)              v = Ub[n];
        else if (n < ADIM + PDIM)  v = betab[n - ADIM];
        biasU[n] = v;
    }
}

// ---------------------------------------------------------------------------
// Ws[b,p,d] = sum_ch parts[b,p,ch]*Ww[d,ch] + Wb[d]   (parts[b,p,ch]=img[b,ch,p])
// ---------------------------------------------------------------------------
__global__ void k_ws(const float* __restrict__ img, const float* __restrict__ Ww,
                     const float* __restrict__ Wb, float* __restrict__ Ws)
{
    const int b = blockIdx.x >> 6;
    const int p = blockIdx.x & 63;
    const int d = threadIdx.x;           // 0..127
    const float* ib = img + (size_t)b * 2048;
    float acc = Wb[d];
#pragma unroll
    for (int ch = 0; ch < 32; ++ch)
        acc += ib[ch * 64 + p] * Ww[d * 32 + ch];
    Ws[((size_t)b * NPARTS + p) * ADIM + d] = acc;
}

// ---------------------------------------------------------------------------
// h0/c0 = tanh(avg @ {ihw,icw}^T + b).  h0 -> Acat[:,0:512] (bf16); also zero pad.
// ---------------------------------------------------------------------------
__global__ void k_init(const float* __restrict__ img,
                       const float* __restrict__ ihw, const float* __restrict__ ihb,
                       const float* __restrict__ icw, const float* __restrict__ icb,
                       __hip_bfloat16* __restrict__ Acat, float* __restrict__ cst)
{
    const int b = blockIdx.x;
    const int n = threadIdx.x;           // 0..511
    __shared__ float savg[64];
    if (n < 64) {
        const float* ib = img + (size_t)b * 2048;
        float s = 0.f;
#pragma unroll
        for (int ch = 0; ch < 32; ++ch) s += ib[ch * 64 + n];
        savg[n] = s * (1.f / 32.f);
    }
    __syncthreads();
    float h0 = ihb[n], c0 = icb[n];
    for (int p = 0; p < 64; ++p) {
        h0 += savg[p] * ihw[n * 64 + p];
        c0 += savg[p] * icw[n * 64 + p];
    }
    Acat[(size_t)b * KCAT + n] = __float2bfloat16(tanhf(h0));
    cst[(size_t)b * HDIM + n]  = tanhf(c0);
    if (n < 20) Acat[(size_t)b * KCAT + 556 + n] = __float2bfloat16(0.f);
}

__device__ __forceinline__ float sigf(float x) { return 1.f / (1.f + expf(-x)); }

// ---------------------------------------------------------------------------
// Attention step: score/softmax/ctx*beta; fills Acat[:,512:556], writes alphas.
// Block = 64 threads (one per part p), grid = BATCH.
// ---------------------------------------------------------------------------
__global__ __launch_bounds__(64)
void k_attn(const float* __restrict__ UB, const float* __restrict__ Ws,
            const float* __restrict__ img, const float* __restrict__ label,
            const float* __restrict__ vw, const float* __restrict__ vb,
            __hip_bfloat16* __restrict__ Acat, float* __restrict__ alphas, int t)
{
    const int b = blockIdx.x;
    const int p = threadIdx.x;           // 0..63
    __shared__ float sUh[ADIM];
    __shared__ float sRed[64];
    __shared__ float sAlpha[64];

    sUh[p]      = UB[(size_t)b * NUB + p];
    sUh[p + 64] = UB[(size_t)b * NUB + 64 + p];
    __syncthreads();

    const float* wsr = Ws + ((size_t)b * NPARTS + p) * ADIM;
    float score = vb[0];
    for (int d = 0; d < ADIM; ++d)
        score += tanhf(wsr[d] + sUh[d]) * vw[d];

    sRed[p] = score; __syncthreads();
    for (int s = 32; s > 0; s >>= 1) { if (p < s) sRed[p] = fmaxf(sRed[p], sRed[p + s]); __syncthreads(); }
    const float m = sRed[0]; __syncthreads();
    const float e = expf(score - m);
    sRed[p] = e; __syncthreads();
    for (int s = 32; s > 0; s >>= 1) { if (p < s) sRed[p] += sRed[p + s]; __syncthreads(); }
    const float alpha = e / sRed[0];
    alphas[((size_t)t * BATCH + b) * NPARTS + p] = alpha;
    sAlpha[p] = alpha;
    __syncthreads();

    if (p < PDIM) {                                  // ctx[d] * beta[d] -> Acat
        const int d = p;
        const float* ib = img + (size_t)b * 2048 + d * 64;   // parts[b,q,d]=img[b,d,q]
        float c = 0.f;
        for (int q = 0; q < NPARTS; ++q) c += sAlpha[q] * ib[q];
        c *= sigf(UB[(size_t)b * NUB + ADIM + d]);
        Acat[(size_t)b * KCAT + HDIM + VOCAB + d] = __float2bfloat16(c);
    }
    if (p < VOCAB)                                   // x_t -> Acat
        Acat[(size_t)b * KCAT + HDIM + p] =
            __float2bfloat16(label[((size_t)b * TSTEPS + t) * VOCAB + p]);
}

// ---------------------------------------------------------------------------
// LSTM cell + output head. Block = 512 threads (one per hidden n), grid = BATCH.
// Waves 0..11 each reduce one logit with __shfl_down; thread 0 does log-softmax.
// ---------------------------------------------------------------------------
__global__ __launch_bounds__(512)
void k_cell(const float* __restrict__ gates, float* __restrict__ cst,
            __hip_bfloat16* __restrict__ Acat,
            const float* __restrict__ ow, const float* __restrict__ ob,
            float* __restrict__ out, int t)
{
    const int b = blockIdx.x;
    const int n = threadIdx.x;
    __shared__ float sh[HDIM];
    __shared__ float sOut[VOCAB];
    __shared__ float sLse[1];

    const float* g = gates + (size_t)b * 2048;
    const float ig = g[n], fg = g[HDIM + n], gg = g[2 * HDIM + n], og = g[3 * HDIM + n];
    const float c_old = cst[(size_t)b * HDIM + n];
    const float c2 = sigf(fg) * c_old + sigf(ig) * tanhf(gg);
    const float h2 = sigf(og) * tanhf(c2);
    cst[(size_t)b * HDIM + n] = c2;
    Acat[(size_t)b * KCAT + n] = __float2bfloat16(h2);   // h for next step's GEMMs
    sh[n] = h2;
    __syncthreads();

    const int wave = n >> 5, lane = n & 31;
    if (wave < VOCAB) {
        float part = 0.f;
        const float* wr = ow + wave * HDIM;
        for (int j = lane; j < HDIM; j += 32) part += sh[j] * wr[j];
#pragma unroll
        for (int off = 16; off > 0; off >>= 1) part += __shfl_down(part, off);
        if (lane == 0) sOut[wave] = part + ob[wave];
    }
    __syncthreads();
    if (n == 0) {
        float m = sOut[0];
        for (int v = 1; v < VOCAB; ++v) m = fmaxf(m, sOut[v]);
        float s = 0.f;
        for (int v = 0; v < VOCAB; ++v) s += expf(sOut[v] - m);
        sLse[0] = m + logf(s);
    }
    __syncthreads();
    if (n < VOCAB)
        out[((size_t)b * TSTEPS + t) * VOCAB + n] = sOut[n] - sLse[0];
}

// ---------------------------------------------------------------------------
// Host-side orchestration (graph-capture safe: only kernel launches on stream)
// ---------------------------------------------------------------------------
extern "C" void kernel_launch(void* const* d_in, const int* in_sizes, int n_in,
                              void* d_out, int out_size, void* d_ws, size_t ws_size,
                              hipStream_t stream)
{
    (void)in_sizes; (void)n_in; (void)out_size; (void)ws_size;
    const float* img   = (const float*)d_in[0];
    const float* label = (const float*)d_in[1];
    const float* Ww    = (const float*)d_in[2];
    const float* Wb    = (const float*)d_in[3];
    const float* Uw    = (const float*)d_in[4];
    const float* Ub    = (const float*)d_in[5];
    const float* vw    = (const float*)d_in[6];
    const float* vb    = (const float*)d_in[7];
    const float* betaw = (const float*)d_in[8];
    const float* betab = (const float*)d_in[9];
    const float* ihw   = (const float*)d_in[10];
    const float* ihb   = (const float*)d_in[11];
    const float* icw   = (const float*)d_in[12];
    const float* icb   = (const float*)d_in[13];
    const float* W_ih  = (const float*)d_in[14];
    const float* W_hh  = (const float*)d_in[15];
    const float* b_ih  = (const float*)d_in[16];
    const float* b_hh  = (const float*)d_in[17];
    const float* ow    = (const float*)d_in[18];
    const float* ob    = (const float*)d_in[19];

    float* out    = (float*)d_out;                               // [B,T,12]
    float* alphas = out + (size_t)BATCH * TSTEPS * VOCAB;        // [T,B,64]

    // Workspace layout (256B aligned)
    char* ws = (char*)d_ws;
    __hip_bfloat16* Acat = (__hip_bfloat16*)(ws);                       // 2048*576*2
    __hip_bfloat16* Wcat = (__hip_bfloat16*)(ws + 2359296);             // 2048*576*2
    __hip_bfloat16* WUB  = (__hip_bfloat16*)(ws + 4718592);             // 192*512*2
    float* biasG = (float*)(ws + 4915200);                              // 2048*4
    float* biasU = (float*)(ws + 4923392);                              // 192*4
    float* cst   = (float*)(ws + 4924416);                              // 2048*512*4
    float* UB    = (float*)(ws + 9118720);                              // 2048*192*4
    float* gates = (float*)(ws + 10691584);                             // 2048*2048*4
    float* Ws    = (float*)(ws + 27468800);                             // 2048*64*128*4
    // total = 94,577,664 bytes

    // --- one-time prep ---
    {
        const int total = 2048 * KCAT + NUB * HDIM + 2048 + NUB;
        k_prep<<<(total + 255) / 256, 256, 0, stream>>>(
            W_hh, W_ih, Uw, betaw, b_ih, b_hh, Ub, betab, Wcat, WUB, biasG, biasU);
        k_ws<<<BATCH * NPARTS, ADIM, 0, stream>>>(img, Ww, Wb, Ws);
        k_init<<<BATCH, HDIM, 0, stream>>>(img, ihw, ihb, icw, icb, Acat, cst);
    }

    // --- sequential decode: 256 steps ---
    const dim3 gemm_blk(256);
    const dim3 gridUB(BATCH / 256, NUB / 64);      // (8, 3)
    const dim3 gridG (BATCH / 256, 2048 / 64);     // (8, 32)
    for (int t = 0; t < TSTEPS; ++t) {
        // UB = h @ [Uw;betaw]^T   (reads Acat[:,0:512] as bf16 h, lda=KCAT)
        gemm_bf16_nt<<<gridUB, gemm_blk, 0, stream>>>(Acat, WUB, biasU, UB,
                                                      HDIM, KCAT, HDIM, NUB);
        // attention -> Acat[:,512:556], alphas
        k_attn<<<BATCH, 64, 0, stream>>>(UB, Ws, img, label, vw, vb, Acat, alphas, t);
        // gates = [h|x|ctx] @ [W_hh|W_ih]^T + (b_ih+b_hh)
        gemm_bf16_nt<<<gridG, gemm_blk, 0, stream>>>(Acat, Wcat, biasG, gates,
                                                     KCAT, KCAT, KCAT, 2048);
        // LSTM cell + logits + log-softmax; writes next h into Acat[:,0:512]
        k_cell<<<BATCH, HDIM, 0, stream>>>(gates, cst, Acat, ow, ob, out, t);
    }
}
